// MI3Graph_71004399337501
// MI455X (gfx1250) — matrix operand order repaired and checked by reading, not compiled, not run on hardware
//
#include <hip/hip_runtime.h>
#include <math.h>

// ---------------------------------------------------------------------------
// MI455X (gfx1250) implementation of the GraphRec-style reference.
//  - Dense [N,64]@[64,64] matmuls -> v_wmma_f32_16x16x32_f16; the f16 weight
//    matrix (8KB) is staged into LDS via async global->LDS B128 copies
//    (ASYNCcnt) instead of a VGPR round-trip.
//  - Edge scatter-adds -> float4 gathers + global_atomic_add_f32.
//  - GAT edge softmax -> monotonic-uint atomicMax trick + atomic denom.
//  - Losses -> wave32 shuffle + LDS block reductions into accumulators.
// ---------------------------------------------------------------------------

#define NUQ 50000
#define NIQ 50000
#define EQ  1000000
#define ETQ 400000

typedef __attribute__((ext_vector_type(16))) _Float16 v16h;
typedef __attribute__((ext_vector_type(8)))  float    v8f;

__device__ __forceinline__ float lrelu(float x, float s) { return x >= 0.f ? x : s * x; }

// monotonic float<->uint encoding for atomic max on arbitrary-sign floats
__device__ __forceinline__ unsigned fenc(float v) {
  unsigned u = __float_as_uint(v);
  return (v >= 0.f) ? (u | 0x80000000u) : ~u;
}
__device__ __forceinline__ float fdec(unsigned u) {
  return (u & 0x80000000u) ? __uint_as_float(u & 0x7FFFFFFFu) : __uint_as_float(~u);
}

// block-wide sum (blockDim.x == 256, 8 wave32s); valid in thread 0
__device__ __forceinline__ float block_sum(float v, float* sbuf) {
  int lane = threadIdx.x & 31, w = threadIdx.x >> 5;
#pragma unroll
  for (int o = 16; o > 0; o >>= 1) v += __shfl_down(v, o, 32);
  if (lane == 0) sbuf[w] = v;
  __syncthreads();
  float r = 0.f;
  if (w == 0) {
    r = (lane < 8) ? sbuf[lane] : 0.f;
#pragma unroll
    for (int o = 16; o > 0; o >>= 1) r += __shfl_down(r, o, 32);
  }
  __syncthreads();
  return r;
}

// convert one 64x64 f32 weight matrix to f16 (for WMMA + async LDS staging)
__global__ void w_to_f16_kernel(const float* __restrict__ W, _Float16* __restrict__ out) {
  int i = blockIdx.x * blockDim.x + threadIdx.x;
  if (i < 64 * 64) out[i] = (_Float16)W[i];
}

// ---------------------------------------------------------------------------
// WMMA GEMM: Y[N,64] = act( (rowscale[m] * X[m,:]) @ W[64,64] + bias )
// act: 0 = none, 1 = LeakyReLU(0.01). rowscale/bias may be null.
// 256 threads = 8 waves, each wave owns 16 rows -> 128 rows per block.
// Wh_g is the f16 weight matrix in global memory; staged into LDS with
// global_load_async_to_lds_b128 (ASYNCcnt) + s_wait_asynccnt.
// ---------------------------------------------------------------------------
__global__ __launch_bounds__(256) void gemm64_kernel(
    const float* __restrict__ X, const _Float16* __restrict__ Wh_g,
    const float* __restrict__ bias, const float* __restrict__ rowscale,
    float* __restrict__ Y, int N, int act) {
  __shared__ __align__(16) _Float16 Wh[64 * 64];  // 8KB
  const int tid = threadIdx.x;

  // async stage: 256 threads x 2 x 16B = 8KB global -> LDS, no VGPR data
  {
    unsigned ldsbase = (unsigned)(uintptr_t)(void*)Wh;
#pragma unroll
    for (int it = 0; it < 2; ++it) {
      int idx = (it * 256 + tid) * 8;  // element (half) index, 16B granules
      unsigned ldso = ldsbase + (unsigned)(idx * 2);
      unsigned long long ga = (unsigned long long)(uintptr_t)(Wh_g + idx);
      asm volatile("global_load_async_to_lds_b128 %0, %1, off"
                   :: "v"(ldso), "v"(ga) : "memory");
    }
    asm volatile("s_wait_asynccnt 0" ::: "memory");
  }
  __syncthreads();

  const int wave = tid >> 5, lane = tid & 31;
  const int half = lane >> 4, lrow = lane & 15;
  const int rowBase = blockIdx.x * 128 + wave * 16;
  const int m = rowBase + lrow;
  const bool valid = (m < N);
  const float s = valid ? (rowscale ? rowscale[m] : 1.0f) : 0.0f;
  const float* xr = X + (size_t)(valid ? m : 0) * 64;
  const bool fullTile = (rowBase + 16 <= N);

  // A fragments (16x32 f16, documented CDNA5 layout): K pairs per VGPR,
  // lanes 0-15 take K {0..7,16..23}+pairs, lanes 16-31 take {8..15,24..31}.
  v16h a0, a1;
#pragma unroll
  for (int v = 0; v < 8; ++v) {
    const int kb = (v < 4 ? 2 * v : 16 + 2 * (v - 4)) + 8 * half;
    a0[2 * v]     = (_Float16)(xr[kb] * s);
    a0[2 * v + 1] = (_Float16)(xr[kb + 1] * s);
    a1[2 * v]     = (_Float16)(xr[kb + 32] * s);
    a1[2 * v + 1] = (_Float16)(xr[kb + 33] * s);
  }

  const int kh = 8 * half;  // B: column n = lrow, K in same table order as A
  for (int tN = 0; tN < 4; ++tN) {
    const int n0 = tN * 16;
    v16h b0, b1;
#pragma unroll
    for (int v = 0; v < 8; ++v) {
      const int kb = (v < 4 ? 2 * v : 16 + 2 * (v - 4)) + kh;
      b0[2 * v]     = Wh[kb * 64 + n0 + lrow];
      b0[2 * v + 1] = Wh[(kb + 1) * 64 + n0 + lrow];
      b1[2 * v]     = Wh[(kb + 32) * 64 + n0 + lrow];
      b1[2 * v + 1] = Wh[(kb + 33) * 64 + n0 + lrow];
    }
    v8f c = {0.f, 0.f, 0.f, 0.f, 0.f, 0.f, 0.f, 0.f};
    c = __builtin_amdgcn_wmma_f32_16x16x32_f16(false, a0, false, b0, (short)0, c, false, false);
    c = __builtin_amdgcn_wmma_f32_16x16x32_f16(false, a1, false, b1, (short)0, c, false, false);
    const int n = n0 + lrow;
    const float bn = bias ? bias[n] : 0.0f;
    if (fullTile) {
      // branch-free fast path (all blocks except the last)
#pragma unroll
      for (int r = 0; r < 8; ++r) {
        const int mr = rowBase + r + 8 * half;
        float yv = c[r] + bn;
        if (act == 1) yv = lrelu(yv, 0.01f);
        Y[(size_t)mr * 64 + n] = yv;
      }
    } else {
#pragma unroll
      for (int r = 0; r < 8; ++r) {
        const int mr = rowBase + r + 8 * half;
        if (mr < N) {
          float yv = c[r] + bn;
          if (act == 1) yv = lrelu(yv, 0.01f);
          Y[(size_t)mr * 64 + n] = yv;
        }
      }
    }
  }
}

// ---------------------------------------------------------------------------
// degree count + rsqrt scaling
// ---------------------------------------------------------------------------
__global__ void count_deg_kernel(const int* __restrict__ idx, float* __restrict__ deg, int n) {
  int i = blockIdx.x * blockDim.x + threadIdx.x;
  if (i < n) atomicAdd(&deg[idx[i]], 1.0f);
}
__global__ void rsqrt_deg_kernel(const float* __restrict__ deg, float* __restrict__ out, int n, float addv) {
  int i = blockIdx.x * blockDim.x + threadIdx.x;
  if (i < n) out[i] = rsqrtf(fmaxf(deg[i] + addv, 1.0f));
}

// agg[dst[e], :] += sSrc[src[e]] * X[src[e], :]   (16 threads/edge, float4 each)
__global__ void edge_scatter_kernel(const float* __restrict__ X, const float* __restrict__ sSrc,
                                    const int* __restrict__ src, const int* __restrict__ dst,
                                    float* __restrict__ agg, int nE) {
  long long t = blockIdx.x * (long long)blockDim.x + threadIdx.x;
  int e = (int)(t >> 4);
  if (e >= nE) return;
  int q = ((int)t & 15) * 4;
  int s = src[e], d = dst[e];
  float sc = sSrc[s];
  float4 x = *(const float4*)(X + (size_t)s * 64 + q);
  float* o = agg + (size_t)d * 64 + q;
  atomicAdd(o + 0, x.x * sc);
  atomicAdd(o + 1, x.y * sc);
  atomicAdd(o + 2, x.z * sc);
  atomicAdd(o + 3, x.w * sc);
}

// agg[i,k] += s[i] * X[i,k]  (self loops; plain RMW, single writer per elem)
__global__ void self_scale_add_kernel(const float* __restrict__ X, const float* __restrict__ s,
                                      float* __restrict__ agg, int n) {
  int idx = blockIdx.x * blockDim.x + threadIdx.x;
  if (idx >= n * 64) return;
  int node = idx >> 6;
  agg[idx] += X[idx] * s[node];
}

__global__ void avg_inplace_kernel(float* __restrict__ a, const float* __restrict__ b, int n) {
  int i = blockIdx.x * blockDim.x + threadIdx.x;
  if (i < n) a[i] = 0.5f * (a[i] + b[i]);
}
__global__ void max_inplace_kernel(float* __restrict__ a, const float* __restrict__ b, int n) {
  int i = blockIdx.x * blockDim.x + threadIdx.x;
  if (i < n) a[i] = fmaxf(a[i], b[i]);
}

// out[i] = dot(H[i,:], v[0:64])
__global__ void rowdot_kernel(const float* __restrict__ H, const float* __restrict__ v,
                              float* __restrict__ out, int N) {
  int i = blockIdx.x * blockDim.x + threadIdx.x;
  if (i >= N) return;
  const float* r = H + (size_t)i * 64;
  float acc = 0.f;
#pragma unroll
  for (int k = 0; k < 64; k += 4) {
    float4 a = *(const float4*)(r + k);
    float4 b = *(const float4*)(v + k);
    acc += a.x * b.x + a.y * b.y + a.z * b.z + a.w * b.w;
  }
  out[i] = acc;
}

// GAT logits over ET real edges + NU self loops; also atomic segment-max
__global__ void gat_logits_kernel(const float* __restrict__ el, const float* __restrict__ er,
                                  const int* __restrict__ ts, const int* __restrict__ td,
                                  float* __restrict__ eE, unsigned* __restrict__ emax,
                                  int nE, int nSelf) {
  int i = blockIdx.x * blockDim.x + threadIdx.x;
  if (i >= nE + nSelf) return;
  int s = (i < nE) ? ts[i] : (i - nE);
  int d = (i < nE) ? td[i] : (i - nE);
  float v = lrelu(el[s] + er[d], 0.2f);
  eE[i] = v;
  atomicMax(&emax[d], fenc(v));
}

__global__ void gat_w_kernel(const float* __restrict__ eE, const unsigned* __restrict__ emax,
                             const int* __restrict__ td, float* __restrict__ wE,
                             float* __restrict__ denom, int nE, int nSelf) {
  int i = blockIdx.x * blockDim.x + threadIdx.x;
  if (i >= nE + nSelf) return;
  int d = (i < nE) ? td[i] : (i - nE);
  float w = expf(eE[i] - fdec(emax[d]));
  wE[i] = w;
  atomicAdd(&denom[d], w);
}

// tAgg[td[e], :] += (w[e]/denom[td[e]]) * h[ts[e], :]
__global__ void gat_scatter_kernel(const float* __restrict__ H, const float* __restrict__ wE,
                                   const float* __restrict__ denom, const int* __restrict__ ts,
                                   const int* __restrict__ td, float* __restrict__ tAgg, int nE) {
  long long t = blockIdx.x * (long long)blockDim.x + threadIdx.x;
  int e = (int)(t >> 4);
  if (e >= nE) return;
  int q = ((int)t & 15) * 4;
  int s = ts[e], d = td[e];
  float alpha = wE[e] / denom[d];
  float4 h = *(const float4*)(H + (size_t)s * 64 + q);
  float* o = tAgg + (size_t)d * 64 + q;
  atomicAdd(o + 0, h.x * alpha);
  atomicAdd(o + 1, h.y * alpha);
  atomicAdd(o + 2, h.z * alpha);
  atomicAdd(o + 3, h.w * alpha);
}

__global__ void gat_self_kernel(const float* __restrict__ H, const float* __restrict__ wSelf,
                                const float* __restrict__ denom, float* __restrict__ tAgg, int n) {
  int idx = blockIdx.x * blockDim.x + threadIdx.x;
  if (idx >= n * 64) return;
  int node = idx >> 6;
  tAgg[idx] += (wSelf[node] / denom[node]) * H[idx];
}

__global__ void bias_lrelu_kernel(const float* __restrict__ X, const float* __restrict__ b,
                                  float* __restrict__ Y, int n) {
  int idx = blockIdx.x * blockDim.x + threadIdx.x;
  if (idx >= n * 64) return;
  Y[idx] = lrelu(X[idx] + b[idx & 63], 0.01f);
}

// per-rated-edge: pos_score, rating loss / mae, attitude loss
__global__ __launch_bounds__(256) void rating_kernel(
    const float* __restrict__ xu2, const float* __restrict__ xi2,
    const float* __restrict__ au, const float* __restrict__ ai,
    const float* __restrict__ ratings, const int* __restrict__ rs, const int* __restrict__ rd,
    float* __restrict__ posOut, float* __restrict__ acc, int nE) {
  __shared__ float sbuf[8];
  int e = blockIdx.x * blockDim.x + threadIdx.x;
  bool active = (e < nE);
  int ec = active ? e : (nE - 1);
  int u = rs[ec], it = rd[ec];
  const float* pa = xu2 + (size_t)u * 64;
  const float* pb = xi2 + (size_t)it * 64;
  const float* pc = au + (size_t)u * 64;
  const float* pd = ai + (size_t)it * 64;
  float p = 0.f, q = 0.f;
#pragma unroll
  for (int k = 0; k < 64; k += 4) {
    float4 a = *(const float4*)(pa + k), b = *(const float4*)(pb + k);
    float4 c = *(const float4*)(pc + k), d = *(const float4*)(pd + k);
    p += a.x * b.x + a.y * b.y + a.z * b.z + a.w * b.w;
    q += c.x * d.x + c.y * d.y + c.z * d.z + c.w * d.w;
  }
  float rv = ratings[ec];
  float diff = rv - p;
  float att = 1.f / (1.f + expf(-q));
  float da = att - rv;
  if (active) posOut[e] = p;
  float rl = block_sum(active ? diff * diff : 0.f, sbuf);
  float ma = block_sum(active ? fabsf(diff) : 0.f, sbuf);
  float la = block_sum(active ? da * da : 0.f, sbuf);
  if (threadIdx.x == 0) {
    atomicAdd(&acc[0], rl);
    atomicAdd(&acc[1], ma);
    atomicAdd(&acc[2], la);
  }
}

__device__ __forceinline__ float softplusf(float x) {
  return fmaxf(x, 0.f) + log1pf(expf(-fabsf(x)));
}

__global__ __launch_bounds__(256) void trust_loss_kernel(
    const float* __restrict__ T, const int* __restrict__ ts, const int* __restrict__ td,
    const int* __restrict__ ns, const int* __restrict__ nd, float* __restrict__ acc, int nE) {
  __shared__ float sbuf[8];
  int e = blockIdx.x * blockDim.x + threadIdx.x;
  bool active = (e < nE);
  int ec = active ? e : (nE - 1);
  const float* a = T + (size_t)ts[ec] * 64;
  const float* b = T + (size_t)td[ec] * 64;
  const float* c = T + (size_t)ns[ec] * 64;
  const float* d = T + (size_t)nd[ec] * 64;
  float p = 0.f, q = 0.f;
#pragma unroll
  for (int k = 0; k < 64; k += 4) {
    float4 av = *(const float4*)(a + k), bv = *(const float4*)(b + k);
    float4 cv = *(const float4*)(c + k), dv = *(const float4*)(d + k);
    p += av.x * bv.x + av.y * bv.y + av.z * bv.z + av.w * bv.w;
    q += cv.x * dv.x + cv.y * dv.y + cv.z * dv.z + cv.w * dv.w;
  }
  float sp = block_sum(active ? softplusf(-p) : 0.f, sbuf);
  float sn = block_sum(active ? softplusf(q) : 0.f, sbuf);
  if (threadIdx.x == 0) {
    atomicAdd(&acc[3], sp);
    atomicAdd(&acc[4], sn);
  }
}

__global__ __launch_bounds__(256) void abs_sum_kernel(const float* __restrict__ X, long long n,
                                                      float* __restrict__ acc) {
  __shared__ float sbuf[8];
  float v = 0.f;
  for (long long i = blockIdx.x * (long long)blockDim.x + threadIdx.x; i < n;
       i += (long long)gridDim.x * blockDim.x)
    v += fabsf(X[i]);
  float r = block_sum(v, sbuf);
  if (threadIdx.x == 0) atomicAdd(acc, r);
}

__global__ __launch_bounds__(256) void abs_diff_sum_kernel(const float* __restrict__ A,
                                                           const float* __restrict__ B,
                                                           long long n, float* __restrict__ acc) {
  __shared__ float sbuf[8];
  float v = 0.f;
  for (long long i = blockIdx.x * (long long)blockDim.x + threadIdx.x; i < n;
       i += (long long)gridDim.x * blockDim.x)
    v += fabsf(A[i] - B[i]);
  float r = block_sum(v, sbuf);
  if (threadIdx.x == 0) atomicAdd(acc, r);
}

__global__ void finalize_kernel(const float* __restrict__ acc, float* __restrict__ out, int E_, int ET_) {
  out[0] = acc[0] / (float)E_;          // rating_loss
  out[1] = acc[1] / (float)E_;          // mae
  out[2] = acc[5];                      // loss_reg (L1 sums)
  out[3 + E_] = acc[2] / (float)E_;     // l_att
  out[4 + E_] = acc[6];                 // loss_a_x
  out[5 + E_] = (acc[3] + acc[4]) / (float)ET_;  // loss_trust
}

// ---------------------------------------------------------------------------
// host-side orchestration
// ---------------------------------------------------------------------------
static void run_gemm(const float* X, const float* W, const float* bias, const float* rowscale,
                     float* Y, int N, int act, _Float16* whbuf, hipStream_t stream) {
  w_to_f16_kernel<<<16, 256, 0, stream>>>(W, whbuf);
  gemm64_kernel<<<(N + 127) / 128, 256, 0, stream>>>(X, whbuf, bias, rowscale, Y, N, act);
}

static void run_conv(const float* X, const int* srcI, const int* dstI, int nE,
                     const float* sSrc, const float* sDst, const float* W, const float* b,
                     float* Y, int nDst, float* agg, _Float16* whbuf, hipStream_t stream) {
  hipMemsetAsync(agg, 0, (size_t)nDst * 64 * sizeof(float), stream);
  long long tE = (long long)nE * 16;
  edge_scatter_kernel<<<(int)((tE + 255) / 256), 256, 0, stream>>>(X, sSrc, srcI, dstI, agg, nE);
  run_gemm(agg, W, b, sDst, Y, nDst, 1, whbuf, stream);
}

extern "C" void kernel_launch(void* const* d_in, const int* in_sizes, int n_in,
                              void* d_out, int out_size, void* d_ws, size_t ws_size,
                              hipStream_t stream) {
  const int NU_ = NUQ, NI_ = NIQ, E_ = EQ, ET_ = ETQ;
  const size_t NF = (size_t)NU_ * 64;

  const float* user_emb  = (const float*)d_in[0];
  const float* item_emb  = (const float*)d_in[1];
  const float* trust_emb = (const float*)d_in[2];
  const float* a_emb_u   = (const float*)d_in[3];
  const float* a_emb_i   = (const float*)d_in[4];
  const float* ratings   = (const float*)d_in[5];
  const float* W1 = (const float*)d_in[6];   const float* b1 = (const float*)d_in[7];
  const float* W2 = (const float*)d_in[8];   const float* b2 = (const float*)d_in[9];
  const float* Wa1 = (const float*)d_in[10]; const float* ba1 = (const float*)d_in[11];
  const float* Wa2 = (const float*)d_in[12]; const float* ba2 = (const float*)d_in[13];
  const float* Wg = (const float*)d_in[14];  const float* bg = (const float*)d_in[15];
  const float* al = (const float*)d_in[16];  const float* ar = (const float*)d_in[17];
  const float* Wt2 = (const float*)d_in[18]; const float* bt2 = (const float*)d_in[19];
  const float* Wau = (const float*)d_in[20]; const float* bau = (const float*)d_in[21];
  const float* Wai = (const float*)d_in[22]; const float* bai = (const float*)d_in[23];
  const int* rs = (const int*)d_in[24];
  const int* rd = (const int*)d_in[25];
  const int* ts = (const int*)d_in[26];
  const int* td = (const int*)d_in[27];
  const int* ns = (const int*)d_in[28];
  const int* nd = (const int*)d_in[29];
  float* out = (float*)d_out;

  // --- workspace carve-out ---
  float* ws = (float*)d_ws;
  size_t off = 0;
  auto alloc = [&](size_t n) { float* p = ws + off; off += (n + 15) & ~(size_t)15; return p; };
  float* agg  = alloc(NF);
  float* x_i  = alloc(NF);
  float* x_u  = alloc(NF);
  float* tmpA = alloc(NF);
  float* x_i2 = alloc(NF);
  float* x_u2 = alloc(NF);
  float* a_i  = alloc(NF);
  float* a_u  = alloc(NF);
  float* a_i2 = alloc(NF);
  float* a_u2 = alloc(NF);
  float* hT   = alloc(NF);
  float* tB   = alloc(NF);
  float* t2   = alloc(NF);
  float* auB  = alloc(NF);
  float* aiB  = alloc(NF);
  float* dru = alloc(NU_); float* dri = alloc(NI_);
  float* dts = alloc(NU_); float* dtd = alloc(NU_);
  float* sru = alloc(NU_); float* sri = alloc(NI_);
  float* sts = alloc(NU_); float* stdv = alloc(NU_);
  float* sgs = alloc(NU_); float* sgd = alloc(NU_);
  float* el = alloc(NU_);  float* er = alloc(NU_);
  float* denom = alloc(NU_);
  unsigned* emax = (unsigned*)alloc(NU_);
  float* eE = alloc(ET_ + NU_);
  float* wE = alloc(ET_ + NU_);
  float* acc = alloc(16);
  _Float16* whbuf = (_Float16*)alloc(2048);  // 4096 halves = 8KB

  auto g1 = [](int n) { return (n + 255) / 256; };

  // --- degrees + normalizers ---
  hipMemsetAsync(dru, 0, NU_ * sizeof(float), stream);
  hipMemsetAsync(dri, 0, NI_ * sizeof(float), stream);
  hipMemsetAsync(dts, 0, NU_ * sizeof(float), stream);
  hipMemsetAsync(dtd, 0, NU_ * sizeof(float), stream);
  count_deg_kernel<<<g1(E_), 256, 0, stream>>>(rs, dru, E_);
  count_deg_kernel<<<g1(E_), 256, 0, stream>>>(rd, dri, E_);
  count_deg_kernel<<<g1(ET_), 256, 0, stream>>>(ts, dts, ET_);
  count_deg_kernel<<<g1(ET_), 256, 0, stream>>>(td, dtd, ET_);
  rsqrt_deg_kernel<<<g1(NU_), 256, 0, stream>>>(dru, sru, NU_, 0.f);
  rsqrt_deg_kernel<<<g1(NI_), 256, 0, stream>>>(dri, sri, NI_, 0.f);
  rsqrt_deg_kernel<<<g1(NU_), 256, 0, stream>>>(dts, sts, NU_, 0.f);
  rsqrt_deg_kernel<<<g1(NU_), 256, 0, stream>>>(dtd, stdv, NU_, 0.f);
  rsqrt_deg_kernel<<<g1(NU_), 256, 0, stream>>>(dts, sgs, NU_, 1.f);  // + self loop
  rsqrt_deg_kernel<<<g1(NU_), 256, 0, stream>>>(dtd, sgd, NU_, 1.f);

  // --- conv1 (mean over relations hitting users) ---
  run_conv(user_emb, rs, rd, E_, sru, sri, W1 + 0,    b1 + 0,   x_i,  NI_, agg, whbuf, stream);
  run_conv(item_emb, rd, rs, E_, sri, sru, W1 + 4096, b1 + 64,  x_u,  NU_, agg, whbuf, stream);
  run_conv(user_emb, ts, td, ET_, sts, stdv, W1 + 8192, b1 + 128, tmpA, NU_, agg, whbuf, stream);
  avg_inplace_kernel<<<g1((int)NF), 256, 0, stream>>>(x_u, tmpA, (int)NF);

  // --- conv2 (max aggregate for users) ---
  run_conv(x_u, rs, rd, E_, sru, sri, W2 + 0,    b2 + 0,   x_i2, NI_, agg, whbuf, stream);
  run_conv(x_i, rd, rs, E_, sri, sru, W2 + 4096, b2 + 64,  x_u2, NU_, agg, whbuf, stream);
  run_conv(x_u, ts, td, ET_, sts, stdv, W2 + 8192, b2 + 128, tmpA, NU_, agg, whbuf, stream);
  max_inplace_kernel<<<g1((int)NF), 256, 0, stream>>>(x_u2, tmpA, (int)NF);

  // --- attitude convs ---
  run_conv(a_emb_u, rs, rd, E_, sru, sri, Wa1 + 0,    ba1 + 0,  a_i,  NI_, agg, whbuf, stream);
  run_conv(a_emb_i, rd, rs, E_, sri, sru, Wa1 + 4096, ba1 + 64, a_u,  NU_, agg, whbuf, stream);
  run_conv(a_u, rs, rd, E_, sru, sri, Wa2 + 0,    ba2 + 0,  a_i2, NI_, agg, whbuf, stream);
  run_conv(a_i, rd, rs, E_, sri, sru, Wa2 + 4096, ba2 + 64, a_u2, NU_, agg, whbuf, stream);

  // --- trust GAT (1 head) on trust graph + self loops ---
  run_gemm(trust_emb, Wg, nullptr, nullptr, hT, NU_, 0, whbuf, stream);
  rowdot_kernel<<<g1(NU_), 256, 0, stream>>>(hT, al, el, NU_);
  rowdot_kernel<<<g1(NU_), 256, 0, stream>>>(hT, ar, er, NU_);
  hipMemsetAsync(emax, 0, NU_ * sizeof(unsigned), stream);  // 0 < fenc(any finite)
  hipMemsetAsync(denom, 0, NU_ * sizeof(float), stream);
  gat_logits_kernel<<<g1(ET_ + NU_), 256, 0, stream>>>(el, er, ts, td, eE, emax, ET_, NU_);
  gat_w_kernel<<<g1(ET_ + NU_), 256, 0, stream>>>(eE, emax, td, wE, denom, ET_, NU_);
  hipMemsetAsync(agg, 0, NF * sizeof(float), stream);
  {
    long long tE = (long long)ET_ * 16;
    gat_scatter_kernel<<<(int)((tE + 255) / 256), 256, 0, stream>>>(hT, wE, denom, ts, td, agg, ET_);
  }
  gat_self_kernel<<<g1((int)NF), 256, 0, stream>>>(hT, wE + ET_, denom, agg, NU_);
  bias_lrelu_kernel<<<g1((int)NF), 256, 0, stream>>>(agg, bg, tB, NU_);

  // --- trust layer 2: graph_conv on (trust + self loops) ---
  hipMemsetAsync(agg, 0, NF * sizeof(float), stream);
  {
    long long tE = (long long)ET_ * 16;
    edge_scatter_kernel<<<(int)((tE + 255) / 256), 256, 0, stream>>>(tB, sgs, ts, td, agg, ET_);
  }
  self_scale_add_kernel<<<g1((int)NF), 256, 0, stream>>>(tB, sgs, agg, NU_);
  run_gemm(agg, Wt2, bt2, sgd, t2, NU_, 1, whbuf, stream);

  // --- attitude projections ---
  run_gemm(a_u2, Wau, bau, nullptr, auB, NU_, 0, whbuf, stream);
  run_gemm(a_i2, Wai, bai, nullptr, aiB, NI_, 0, whbuf, stream);

  // --- losses ---
  hipMemsetAsync(acc, 0, 16 * sizeof(float), stream);
  rating_kernel<<<g1(E_), 256, 0, stream>>>(x_u2, x_i2, auB, aiB, ratings, rs, rd, out + 3, acc, E_);
  trust_loss_kernel<<<g1(ET_), 256, 0, stream>>>(t2, ts, td, ns, nd, acc, ET_);
  abs_sum_kernel<<<512, 256, 0, stream>>>(x_u2, (long long)NF, acc + 5);
  abs_sum_kernel<<<512, 256, 0, stream>>>(x_i2, (long long)NF, acc + 5);
  abs_sum_kernel<<<512, 256, 0, stream>>>(t2,   (long long)NF, acc + 5);
  abs_sum_kernel<<<512, 256, 0, stream>>>(a_u2, (long long)NF, acc + 5);
  abs_sum_kernel<<<512, 256, 0, stream>>>(a_i2, (long long)NF, acc + 5);
  abs_diff_sum_kernel<<<512, 256, 0, stream>>>(x_i2, a_i2, (long long)NF, acc + 6);
  finalize_kernel<<<1, 1, 0, stream>>>(acc, out, E_, ET_);
}
